// RN_51874615001752
// MI455X (gfx1250) — matrix operand order repaired and checked by reading, not compiled, not run on hardware
//
#include <hip/hip_runtime.h>
#include <stdint.h>

// ---------------- CDNA5 WMMA types ----------------
typedef __attribute__((ext_vector_type(16))) __bf16 v16bf;
typedef __attribute__((ext_vector_type(8)))  float  v8f;

struct alignas(16) U4 { uint32_t x[4]; };
union ABFrag { v16bf v; U4 q[2]; };

__device__ __forceinline__ uint16_t f2bf(float f) {
  uint32_t u = __float_as_uint(f);
  uint32_t r = (u + 0x7FFFu + ((u >> 16) & 1u)) >> 16;
  return (uint16_t)r;
}

#define RN 256
#define PAIRS 2401        // 49*49
#define TILES 151         // ceil(2401/16)
#define CHUNKS 8
#define TILES_PER_CHUNK 19
#define H0_STRIDE 264     // 256 + 8 halves pad (528B row, breaks bank conflicts)

// ---------------- conv 3x3 stride2 pad1 (direct) ----------------
__global__ void k_conv3x3s2(const float* __restrict__ in, const float* __restrict__ w,
                            const float* __restrict__ bias, float* __restrict__ out,
                            int N, int Cin, int Hin, int Win, int Cout, int Hout, int Wout) {
  int idx = blockIdx.x * 256 + threadIdx.x;
  int total = N * Cout * Hout * Wout;
  if (idx >= total) return;
  int wo = idx % Wout; int t = idx / Wout;
  int ho = t % Hout; t /= Hout;
  int co = t % Cout; int n = t / Cout;
  float acc = bias[co];
  int ih0 = ho * 2 - 1, iw0 = wo * 2 - 1;
  for (int ci = 0; ci < Cin; ++ci) {
    const float* ip = in + ((size_t)(n * Cin + ci) * Hin) * Win;
    const float* wp = w + (size_t)(co * Cin + ci) * 9;
#pragma unroll
    for (int kh = 0; kh < 3; ++kh) {
      int ih = ih0 + kh;
      if (ih < 0 || ih >= Hin) continue;
#pragma unroll
      for (int kw = 0; kw < 3; ++kw) {
        int iw = iw0 + kw;
        if (iw < 0 || iw >= Win) continue;
        acc = fmaf(ip[(size_t)ih * Win + iw], wp[kh * 3 + kw], acc);
      }
    }
  }
  out[idx] = acc;
}

// ---------------- BN stats: per-(n,c) partial sums (deterministic) ----------------
__global__ void k_bn_stats(const float* __restrict__ y, float* __restrict__ part, int HW) {
  const float* p = y + (size_t)blockIdx.x * HW;
  float s = 0.f, s2 = 0.f;
  for (int i = threadIdx.x; i < HW; i += 256) { float v = p[i]; s += v; s2 += v * v; }
  __shared__ float rs[256], rs2[256];
  rs[threadIdx.x] = s; rs2[threadIdx.x] = s2;
  __syncthreads();
  for (int off = 128; off > 0; off >>= 1) {
    if (threadIdx.x < off) { rs[threadIdx.x] += rs[threadIdx.x + off]; rs2[threadIdx.x] += rs2[threadIdx.x + off]; }
    __syncthreads();
  }
  if (threadIdx.x == 0) { part[blockIdx.x * 2] = rs[0]; part[blockIdx.x * 2 + 1] = rs2[0]; }
}

// reduce per-(n,c) partials over N=128 images into per-channel sums (fixed order)
__global__ void k_bn_reduce(const float* __restrict__ part, float* __restrict__ stats, int C) {
  int c = blockIdx.x;
  __shared__ float s1[128], s2[128];
  int t = threadIdx.x;
  s1[t] = part[(t * C + c) * 2];
  s2[t] = part[(t * C + c) * 2 + 1];
  __syncthreads();
  for (int off = 64; off > 0; off >>= 1) {
    if (t < off) { s1[t] += s1[t + off]; s2[t] += s2[t + off]; }
    __syncthreads();
  }
  if (t == 0) { stats[c * 2] = s1[0]; stats[c * 2 + 1] = s2[0]; }
}

__global__ void k_bn_apply(float* __restrict__ y, const float* __restrict__ stats,
                           const float* __restrict__ gamma, const float* __restrict__ beta,
                           int C, int HW, int total, float invcnt) {
  int idx = blockIdx.x * 256 + threadIdx.x;
  if (idx >= total) return;
  int c = (idx / HW) % C;
  float mean = stats[c * 2] * invcnt;
  float var  = stats[c * 2 + 1] * invcnt - mean * mean;
  float sc = rsqrtf(var + 1e-5f) * gamma[c];
  y[idx] = (y[idx] - mean) * sc + beta[c];
}

// ---------------- hi/hj projections: x_flat @ w_g1 halves ----------------
// x5: (128,24,7,7); hi[b,s,:] = sum_c x[b,c,s]*w_g1[24+c,:], hj uses w_g1[c,:]
__global__ void k_proj(const float* __restrict__ x5, const float* __restrict__ wg1,
                       float* __restrict__ hi, float* __restrict__ hj) {
  int bs = blockIdx.x;                 // b*49 + s
  int b = bs / 49, s = bs - b * 49;
  __shared__ float xr[24];
  if (threadIdx.x < 24) xr[threadIdx.x] = x5[(size_t)(b * 24 + threadIdx.x) * 49 + s];
  __syncthreads();
  int n = threadIdx.x;
  float a = 0.f, c = 0.f;
#pragma unroll
  for (int k = 0; k < 24; ++k) {
    float xv = xr[k];
    c = fmaf(xv, wg1[k * RN + n], c);          // h_j contribution (rows 0..23)
    a = fmaf(xv, wg1[(24 + k) * RN + n], a);   // h_i contribution (rows 24..47)
  }
  hi[(size_t)bs * RN + n] = a;
  hj[(size_t)bs * RN + n] = c;
}

// ---------------- pre-convert w_g (256x256 f32) to bf16 WMMA B-fragment layout ----
// frag index f = kk*16+nt; dword slot = f*256 + lane*8 + r
// lane<16: N=nt*16+lane, K = kk*32 + 2r,2r+1 (+8 if r>=4) ; lane>=16: K += 8
__global__ void k_prep_wg(const float* __restrict__ wg, uint32_t* __restrict__ wfrag) {
  int idx = blockIdx.x * 256 + threadIdx.x;      // 0..32767
  int r = idx & 7;
  int lane = (idx >> 3) & 31;
  int f = idx >> 8;
  int kk = f >> 4, nt = f & 15;
  int k0 = kk * 32 + ((lane >> 4) << 3) + 2 * r + ((r >> 2) << 3);
  int n  = nt * 16 + (lane & 15);
  uint32_t lo = f2bf(wg[k0 * RN + n]);
  uint32_t hi = f2bf(wg[(k0 + 1) * RN + n]);
  wfrag[idx] = lo | (hi << 16);
}

// ---------------- fused RN core: h0 gen -> g1 -> g2 -> pair-sum ----------------
// block = 256 threads (8 wave32); wave w owns N columns [32w, 32w+32)
// B fragments of w_g (shared by both g layers) live in REGISTERS for the
// whole kernel (16 frags = 128 VGPRs); LDS holds only h0/h1 tiles + staging.
__global__ void __launch_bounds__(256)
k_rn_fused(const float* __restrict__ hi, const float* __restrict__ hj,
           const float* __restrict__ bg1, const uint32_t* __restrict__ wfrag_g,
           const float* __restrict__ bg, float* __restrict__ parts) {
  __shared__ uint16_t h0[16 * H0_STRIDE];   // 8448 B
  __shared__ uint16_t h1[16 * H0_STRIDE];   // 8448 B
  __shared__ float    csum[RN * 2];         // 2048 B

  const int tid = threadIdx.x;
  const int b     = blockIdx.x / CHUNKS;
  const int chunk = blockIdx.x % CHUNKS;
  const int tbeg = chunk * TILES_PER_CHUNK;
  const int tend = (tbeg + TILES_PER_CHUNK < TILES) ? (tbeg + TILES_PER_CHUNK) : TILES;

  const int lane   = tid & 31;
  const int wid    = tid >> 5;
  const int lane15 = lane & 15;
  const int lhalf  = lane >> 4;
  const int nt0 = wid * 2, nt1 = wid * 2 + 1;
  const int n0 = nt0 * 16 + lane15, n1 = nt1 * 16 + lane15;
  const float bb0 = bg[n0], bb1 = bg[n1];
  const float b1col = bg1[tid];
  const float* hib = hi + (size_t)b * 49 * RN;
  const float* hjb = hj + (size_t)b * 49 * RN;
  float cs0 = 0.f, cs1 = 0.f;
  const v8f vz = {0.f, 0.f, 0.f, 0.f, 0.f, 0.f, 0.f, 0.f};

  // ---- load this wave's 16 w_g B-fragments into registers (stay hot in L2) ----
  ABFrag B0[8], B1[8];
#pragma unroll
  for (int kk = 0; kk < 8; ++kk) {
    const U4* g0 = (const U4*)(wfrag_g + ((kk * 16 + nt0) * 256 + lane * 8));
    const U4* g1 = (const U4*)(wfrag_g + ((kk * 16 + nt1) * 256 + lane * 8));
    B0[kk].q[0] = g0[0]; B0[kk].q[1] = g0[1];
    B1[kk].q[0] = g1[0]; B1[kk].q[1] = g1[1];
  }

  for (int t = tbeg; t < tend; ++t) {
    const int pbase = t * 16;
    // --- h0 tile generation: 16 pair-rows x 256 cols, bf16, A-friendly layout ---
#pragma unroll
    for (int m = 0; m < 16; ++m) {
      int p = pbase + m;
      float v = 0.f;
      if (p < PAIRS) {
        int i = p / 49, j = p - i * 49;
        v = fmaxf(hib[i * RN + tid] + hjb[j * RN + tid] + b1col, 0.f);
      }
      h0[m * H0_STRIDE + tid] = f2bf(v);
    }
    __syncthreads();

    // --- layer 1: h1 = relu(h0 @ w_g + b_g) ---
    v8f acc0 = vz, acc1 = vz;
    {
      const char* abase = (const char*)h0 + lane15 * (H0_STRIDE * 2) + lhalf * 16;
#pragma unroll
      for (int kk = 0; kk < 8; ++kk) {
        ABFrag a;
        const char* ap = abase + kk * 64;
        a.q[0] = *(const U4*)ap;
        a.q[1] = *(const U4*)(ap + 32);
        acc0 = __builtin_amdgcn_wmma_f32_16x16x32_bf16(false, a.v, false, B0[kk].v, (short)0, acc0, false, false);
        acc1 = __builtin_amdgcn_wmma_f32_16x16x32_bf16(false, a.v, false, B1[kk].v, (short)0, acc1, false, false);
      }
    }
#pragma unroll
    for (int r = 0; r < 8; ++r) {
      int M = r + 8 * lhalf;                     // C/D layout: VGPR r, lane-half -> row
      h1[M * H0_STRIDE + n0] = f2bf(fmaxf(acc0[r] + bb0, 0.f));
      h1[M * H0_STRIDE + n1] = f2bf(fmaxf(acc1[r] + bb1, 0.f));
    }
    __syncthreads();

    // --- layer 2: h2 = relu(h1 @ w_g + b_g), masked pair-sum into registers ---
    acc0 = vz; acc1 = vz;
    {
      const char* abase = (const char*)h1 + lane15 * (H0_STRIDE * 2) + lhalf * 16;
#pragma unroll
      for (int kk = 0; kk < 8; ++kk) {
        ABFrag a;
        const char* ap = abase + kk * 64;
        a.q[0] = *(const U4*)ap;
        a.q[1] = *(const U4*)(ap + 32);
        acc0 = __builtin_amdgcn_wmma_f32_16x16x32_bf16(false, a.v, false, B0[kk].v, (short)0, acc0, false, false);
        acc1 = __builtin_amdgcn_wmma_f32_16x16x32_bf16(false, a.v, false, B1[kk].v, (short)0, acc1, false, false);
      }
    }
#pragma unroll
    for (int r = 0; r < 8; ++r) {
      int M = r + 8 * lhalf;
      float msk = (pbase + M < PAIRS) ? 1.f : 0.f;
      cs0 += msk * fmaxf(acc0[r] + bb0, 0.f);
      cs1 += msk * fmaxf(acc1[r] + bb1, 0.f);
    }
    __syncthreads();
  }

  // combine the two lane-halves per column, write per-chunk partial (deterministic)
  csum[n0 * 2 + lhalf] = cs0;
  csum[n1 * 2 + lhalf] = cs1;
  __syncthreads();
  parts[(size_t)blockIdx.x * RN + tid] = csum[tid * 2] + csum[tid * 2 + 1];
}

// ---------------- f layers: x = relu(x@w_f+b_f) twice ----------------
__global__ void k_flayers(const float* __restrict__ parts, const float* __restrict__ wf,
                          const float* __restrict__ bf, float* __restrict__ out) {
  __shared__ float row[RN], row2[RN];
  int b = blockIdx.x, n = threadIdx.x;
  float s = 0.f;
#pragma unroll
  for (int c = 0; c < CHUNKS; ++c) s += parts[(size_t)(b * CHUNKS + c) * RN + n];
  row[n] = s;
  __syncthreads();
  float a = bf[n];
  for (int k = 0; k < RN; ++k) a = fmaf(row[k], wf[k * RN + n], a);
  row2[n] = fmaxf(a, 0.f);
  __syncthreads();
  a = bf[n];
  for (int k = 0; k < RN; ++k) a = fmaf(row2[k], wf[k * RN + n], a);
  out[(size_t)b * RN + n] = fmaxf(a, 0.f);
}

// ---------------- launcher ----------------
extern "C" void kernel_launch(void* const* d_in, const int* in_sizes, int n_in,
                              void* d_out, int out_size, void* d_ws, size_t ws_size,
                              hipStream_t stream) {
  const float* img = (const float*)d_in[0];
  const float* w0  = (const float*)d_in[1];
  const float* b0  = (const float*)d_in[2];
  const float* wk  = (const float*)d_in[3];
  const float* bk  = (const float*)d_in[4];
  const float* bng = (const float*)d_in[5];
  const float* bnb = (const float*)d_in[6];
  const float* wg1 = (const float*)d_in[7];
  const float* bg1 = (const float*)d_in[8];
  const float* wg  = (const float*)d_in[9];
  const float* bg  = (const float*)d_in[10];
  const float* wf  = (const float*)d_in[11];
  const float* bf  = (const float*)d_in[12];
  float* out = (float*)d_out;
  char* ws = (char*)d_ws;

  // workspace layout (bytes)
  float* o1 = (float*)(ws + 0ull);                 // 128*24*112*112
  float* o2 = (float*)(ws + 154140672ull);         // 128*24*56*56
  float* o3 = (float*)(ws + 192675840ull);         // 128*24*28*28
  float* o4 = (float*)(ws + 202309632ull);         // 128*24*14*14
  float* o5 = (float*)(ws + 204718080ull);         // 128*24*7*7
  float* hi = (float*)(ws + 205320192ull);         // 128*49*256
  float* hj = (float*)(ws + 211742720ull);         // 128*49*256
  uint32_t* wfrag = (uint32_t*)(ws + 218165248ull);// 32768 dwords
  float* parts  = (float*)(ws + 218296320ull);     // 128*8*256
  float* bnpart = (float*)(ws + 219344896ull);     // 3072*2
  float* bnstat = (float*)(ws + 219369472ull);     // 48

  struct CL { const float* w; const float* b; const float* g; const float* be;
              int cin, hin, win, hout, wout; };
  CL layers[5] = {
    { w0,            b0,      bng +  0, bnb +  0,  3, 224, 224, 112, 112 },
    { wk +  0*5184,  bk +  0, bng + 24, bnb + 24, 24, 112, 112,  56,  56 },
    { wk +  1*5184,  bk + 24, bng + 48, bnb + 48, 24,  56,  56,  28,  28 },
    { wk +  2*5184,  bk + 48, bng + 72, bnb + 72, 24,  28,  28,  14,  14 },
    { wk +  3*5184,  bk + 72, bng + 96, bnb + 96, 24,  14,  14,   7,   7 },
  };
  float* bufs[6] = { (float*)img, o1, o2, o3, o4, o5 };

  for (int L = 0; L < 5; ++L) {
    const CL& cl = layers[L];
    float* dst = bufs[L + 1];
    int hw = cl.hout * cl.wout;
    int total = 128 * 24 * hw;
    k_conv3x3s2<<<(total + 255) / 256, 256, 0, stream>>>(
        bufs[L], cl.w, cl.b, dst, 128, cl.cin, cl.hin, cl.win, 24, cl.hout, cl.wout);
    k_bn_stats<<<128 * 24, 256, 0, stream>>>(dst, bnpart, hw);
    k_bn_reduce<<<24, 128, 0, stream>>>(bnpart, bnstat, 24);
    k_bn_apply<<<(total + 255) / 256, 256, 0, stream>>>(
        dst, bnstat, cl.g, cl.be, 24, hw, total, 1.0f / (float)(128 * hw));
  }

  k_proj<<<128 * 49, 256, 0, stream>>>(o5, wg1, hi, hj);
  k_prep_wg<<<128, 256, 0, stream>>>(wg, wfrag);
  k_rn_fused<<<128 * CHUNKS, 256, 0, stream>>>(hi, hj, bg1, wfrag, bg, parts);
  k_flayers<<<128, 256, 0, stream>>>(parts, wf, bf, out);

  (void)in_sizes; (void)n_in; (void)out_size; (void)ws_size;
}
